// Block_54004918780441
// MI455X (gfx1250) — compile-verified, matrix operand order checked by reading, not run on hardware
//
#include <hip/hip_runtime.h>
#include <hip/hip_bf16.h>
#include <math.h>

#define DEV __device__ __forceinline__

typedef __attribute__((ext_vector_type(16))) __bf16 v16bf;
typedef __attribute__((ext_vector_type(8)))  float  v8f;

struct alignas(16) Q16 { unsigned u0, u1, u2, u3; };

static constexpr int C_    = 768;
static constexpr int NHD   = 12;
static constexpr int HD    = 64;
static constexpr int WS_   = 14;
static constexpr int NW    = 196;     // tokens per window (14*14)
static constexpr int NWIN  = 200;     // 8 batches * 25 windows
static constexpr int TOKW  = 39200;   // NWIN * NW
static constexpr int TOKI  = 32768;   // 8*64*64
static constexpr int N_QKV = 2304;
static constexpr int N_MLP = 3072;

DEV unsigned short f2bf_bits(float f) {
  union { float f; unsigned u; } x; x.f = f;
  unsigned r = x.u + 0x7fffu + ((x.u >> 16) & 1u);   // RNE
  return (unsigned short)(r >> 16);
}
DEV __bf16 f2bf(float f) {
  unsigned short s = f2bf_bits(f);
  __bf16 b; __builtin_memcpy(&b, &s, 2); return b;
}

DEV void load_frag(v16bf& f, const __bf16* p0, const __bf16* p1) {
  Q16* d = reinterpret_cast<Q16*>(&f);
  d[0] = *reinterpret_cast<const Q16*>(p0);
  d[1] = *reinterpret_cast<const Q16*>(p1);
}
DEV void zero_frag(v16bf& f) {
  Q16 z = {0u, 0u, 0u, 0u};
  Q16* d = reinterpret_cast<Q16*>(&f);
  d[0] = z; d[1] = z;
}
DEV v8f wmma_bf16(v16bf a, v16bf b, v8f c) {
  return __builtin_amdgcn_wmma_f32_16x16x32_bf16(false, a, false, b, (short)0, c,
                                                 false, false);
}

// ---------------- weight transpose + cast: wT[n*K + k] = bf16(w[k*N + n]) ---
__global__ __launch_bounds__(256) void castT_kernel(const float* __restrict__ w,
                                                    __bf16* __restrict__ wT,
                                                    int Kd, int Nd) {
  long i = (long)blockIdx.x * 256 + threadIdx.x;
  long total = (long)Kd * Nd;
  if (i >= total) return;
  long n = i / Kd, k = i - n * Kd;
  wT[i] = f2bf(w[k * (long)Nd + n]);
}

// ---------------- LayerNorm, one wave per row, optional window-order output -
template<bool WINDOWED>
__global__ __launch_bounds__(256) void ln_kernel(const float* __restrict__ x,
                                                 const float* __restrict__ g,
                                                 const float* __restrict__ bta,
                                                 __bf16* __restrict__ out,
                                                 int rows) {
  int wave = threadIdx.x >> 5, lane = threadIdx.x & 31;
  long row = (long)blockIdx.x * 8 + wave;
  if (row >= rows) return;
  const float* src;
  bool valid = true;
  if constexpr (WINDOWED) {
    int win = (int)(row / NW), p = (int)(row % NW);
    int bb = win / 25, wr = win % 25;
    int wy = wr / 5, wx = wr % 5;
    int py = p / WS_, px = p % WS_;
    int yy = wy * WS_ + py, xx = wx * WS_ + px;
    valid = (yy < 64) && (xx < 64);
    src = x + (((long)bb * 64 + yy) * 64 + xx) * C_;
  } else {
    src = x + row * C_;
  }
  __bf16* o = out + row * C_;
  if (!valid) {
    for (int c = lane; c < C_; c += 32) o[c] = f2bf(0.0f);
    return;
  }
  float vals[24];
  float s = 0.f;
  #pragma unroll
  for (int i = 0; i < 24; ++i) { vals[i] = src[lane + i * 32]; s += vals[i]; }
  #pragma unroll
  for (int off = 16; off >= 1; off >>= 1) s += __shfl_xor(s, off, 32);
  float mean = s * (1.0f / C_);
  float vs = 0.f;
  #pragma unroll
  for (int i = 0; i < 24; ++i) { float d = vals[i] - mean; vs += d * d; }
  #pragma unroll
  for (int off = 16; off >= 1; off >>= 1) vs += __shfl_xor(vs, off, 32);
  float rstd = rsqrtf(vs * (1.0f / C_) + 1e-5f);
  #pragma unroll
  for (int i = 0; i < 24; ++i) {
    int c = lane + i * 32;
    o[c] = f2bf((vals[i] - mean) * rstd * g[c] + bta[c]);
  }
}

// ---------------- generic bf16 WMMA GEMM: C = A[M,K] * Bt[N,K]^T ------------
// Wave tile 32x64 (2 A-frags x 4 B-frags -> 8 WMMA/k-step), block tile 64x256.
// MODE 0: bf16 out + bias (QKV)
// MODE 1: f32 scatter window->image + bias + residual gather (proj -> x1)
// MODE 2: bf16 out + bias + erf-GELU (fc1)
// MODE 3: f32 out + bias + row-major residual (fc2 -> d_out)
template<int MODE>
__global__ __launch_bounds__(256) void gemm_kernel(const __bf16* __restrict__ A,
                                                   const __bf16* __restrict__ Bt,
                                                   const float* __restrict__ bias,
                                                   const float* __restrict__ resid,
                                                   void* __restrict__ outp,
                                                   int M, int N, int K) {
  int tid = threadIdx.x;
  int wave = tid >> 5, lane = tid & 31;
  int half = lane >> 4, l = lane & 15;
  int mw = wave >> 2, nw = wave & 3;
  int rowBase = blockIdx.y * 64 + mw * 32;
  int colBase = blockIdx.x * 256 + nw * 64;

  v8f acc[2][4];
  #pragma unroll
  for (int i = 0; i < 2; ++i)
    #pragma unroll
    for (int j = 0; j < 4; ++j) { v8f z = {}; acc[i][j] = z; }

  const int r0 = rowBase + l, r1 = rowBase + 16 + l;
  const __bf16* a0p = A + (long)r0 * K;
  const __bf16* a1p = A + (long)r1 * K;
  const __bf16* bp[4];
  #pragma unroll
  for (int j = 0; j < 4; ++j) bp[j] = Bt + (long)(colBase + j * 16 + l) * K;
  const bool v0 = r0 < M, v1 = r1 < M;
  const int ho = half * 8;

  for (int k0 = 0; k0 < K; k0 += 32) {
    v16bf a0, a1;
    if (v0) load_frag(a0, a0p + k0 + ho, a0p + k0 + 16 + ho); else zero_frag(a0);
    if (v1) load_frag(a1, a1p + k0 + ho, a1p + k0 + 16 + ho); else zero_frag(a1);
    #pragma unroll
    for (int j = 0; j < 4; ++j) {
      v16bf b;
      load_frag(b, bp[j] + k0 + ho, bp[j] + k0 + 16 + ho);
      acc[0][j] = wmma_bf16(a0, b, acc[0][j]);
      acc[1][j] = wmma_bf16(a1, b, acc[1][j]);
    }
  }

  #pragma unroll
  for (int mf = 0; mf < 2; ++mf) {
    #pragma unroll
    for (int r = 0; r < 8; ++r) {
      int row = rowBase + mf * 16 + r + half * 8;
      if (row >= M) continue;
      long sidx = -1;
      if constexpr (MODE == 1) {
        int win = row / NW, p = row % NW;
        int bb = win / 25, wr = win % 25;
        int wy = wr / 5, wx = wr % 5;
        int py = p / WS_, px = p % WS_;
        int yy = wy * WS_ + py, xx = wx * WS_ + px;
        if (yy < 64 && xx < 64) sidx = (((long)bb * 64 + yy) * 64 + xx) * C_;
      }
      #pragma unroll
      for (int nt = 0; nt < 4; ++nt) {
        int col = colBase + nt * 16 + l;
        float v = acc[mf][nt][r] + bias[col];
        if constexpr (MODE == 0) {
          ((__bf16*)outp)[(long)row * N + col] = f2bf(v);
        } else if constexpr (MODE == 2) {
          float gl = 0.5f * v * (1.0f + erff(v * 0.70710678f));
          ((__bf16*)outp)[(long)row * N + col] = f2bf(gl);
        } else if constexpr (MODE == 3) {
          long idx = (long)row * N + col;
          ((float*)outp)[idx] = v + resid[idx];
        } else {  // MODE 1
          if (sidx >= 0) {
            long idx = sidx + col;
            ((float*)outp)[idx] = v + resid[idx];
          }
        }
      }
    }
  }
}

// ---------------- windowed attention: one block per (window, head) ----------
__global__ __launch_bounds__(256) void attn_kernel(const __bf16* __restrict__ qkv,
                                                   __bf16* __restrict__ attn) {
  const int win = blockIdx.x, head = blockIdx.y;
  const long tokBase = (long)win * NW;
  const long rowStride = N_QKV;
  const long qOff = head * HD;
  const long kOff = C_ + head * HD;
  const long vOff = 2 * C_ + head * HD;
  const __bf16* base = qkv + tokBase * rowStride;

  __shared__ __bf16 Ksh[208 * 64];       // K rows (Bt for Q*K^T), pad rows 0
  __shared__ __bf16 Vt[64 * 224];        // V transposed (Bt for P*V), pad cols 0
  __shared__ __bf16 Psh[8 * 16 * 224];   // per-wave P staging

  const int tid = threadIdx.x;
  for (int i = tid; i < 208 * 8; i += 256) {
    int t = i >> 3, c = (i & 7) * 8;
    Q16 d = {0u, 0u, 0u, 0u};
    if (t < NW) d = *reinterpret_cast<const Q16*>(base + (long)t * rowStride + kOff + c);
    *reinterpret_cast<Q16*>(&Ksh[t * 64 + c]) = d;
  }
  for (int i = tid; i < 208 * 8; i += 256) {
    int t = i >> 3, c = (i & 7) * 8;
    if (t < NW) {
      union { Q16 q; __bf16 h[8]; } u;
      u.q = *reinterpret_cast<const Q16*>(base + (long)t * rowStride + vOff + c);
      #pragma unroll
      for (int j = 0; j < 8; ++j) Vt[(c + j) * 224 + t] = u.h[j];
    } else {
      #pragma unroll
      for (int j = 0; j < 8; ++j) Vt[(c + j) * 224 + t] = f2bf(0.0f);
    }
  }
  for (int i = tid; i < 64 * 16; i += 256) {
    int d = i >> 4, t = 208 + (i & 15);
    Vt[d * 224 + t] = f2bf(0.0f);
  }
  __syncthreads();

  const int wave = tid >> 5, lane = tid & 31;
  const int half = lane >> 4, l = lane & 15;
  const int ho = half * 8;
  __bf16* Pw = &Psh[wave * 16 * 224];

  for (int rt = wave; rt < 13; rt += 8) {
    v16bf qf0, qf1;
    {
      int t = rt * 16 + l;
      if (t < NW) {
        const __bf16* qp = base + (long)t * rowStride + qOff;
        load_frag(qf0, qp + ho,      qp + 16 + ho);
        load_frag(qf1, qp + 32 + ho, qp + 48 + ho);
      } else { zero_frag(qf0); zero_frag(qf1); }
    }
    // S = (Q K^T) * scale
    v8f sc[13];
    #pragma unroll
    for (int ct = 0; ct < 13; ++ct) {
      v16bf b0, b1;
      const __bf16* kp = &Ksh[(ct * 16 + l) * 64];
      load_frag(b0, kp + ho,      kp + 16 + ho);
      load_frag(b1, kp + 32 + ho, kp + 48 + ho);
      v8f s = {};
      s = wmma_bf16(qf0, b0, s);
      s = wmma_bf16(qf1, b1, s);
      sc[ct] = s;
    }
    #pragma unroll
    for (int ct = 0; ct < 13; ++ct) {
      bool oob = (ct * 16 + l) >= NW;
      #pragma unroll
      for (int r = 0; r < 8; ++r) {
        float v = sc[ct][r] * 0.125f;   // 1/sqrt(64)
        sc[ct][r] = oob ? -1e30f : v;
      }
    }
    // softmax (rows striped: m = r + 8*half, cols across 16-lane half)
    float rinv[8];
    #pragma unroll
    for (int r = 0; r < 8; ++r) {
      float m = -1e30f;
      #pragma unroll
      for (int ct = 0; ct < 13; ++ct) m = fmaxf(m, sc[ct][r]);
      #pragma unroll
      for (int off = 8; off >= 1; off >>= 1) m = fmaxf(m, __shfl_xor(m, off, 32));
      bool rowValid = (rt * 16 + r + half * 8) < NW;
      float s = 0.f;
      #pragma unroll
      for (int ct = 0; ct < 13; ++ct) {
        float p = rowValid ? __expf(sc[ct][r] - m) : 0.0f;
        sc[ct][r] = p;
        s += p;
      }
      #pragma unroll
      for (int off = 8; off >= 1; off >>= 1) s += __shfl_xor(s, off, 32);
      rinv[r] = (s > 0.f) ? (1.0f / s) : 0.0f;
    }
    // P -> LDS (A-fragment relayout), zero the K-pad columns 208..223
    #pragma unroll
    for (int r = 0; r < 8; ++r) {
      int lm = r + half * 8;
      #pragma unroll
      for (int ct = 0; ct < 13; ++ct)
        Pw[lm * 224 + ct * 16 + l] = f2bf(sc[ct][r]);
      Pw[lm * 224 + 208 + l] = f2bf(0.0f);
    }
    // O = P @ V
    v8f o0 = {}, o1 = {}, o2 = {}, o3 = {};
    #pragma unroll
    for (int ks = 0; ks < 7; ++ks) {
      int k0 = ks * 32;
      v16bf pa, vb;
      const __bf16* pp = Pw + l * 224 + k0;
      load_frag(pa, pp + ho, pp + 16 + ho);
      const __bf16* vp0 = &Vt[(0 * 16 + l) * 224 + k0];
      load_frag(vb, vp0 + ho, vp0 + 16 + ho);
      o0 = wmma_bf16(pa, vb, o0);
      const __bf16* vp1 = &Vt[(1 * 16 + l) * 224 + k0];
      load_frag(vb, vp1 + ho, vp1 + 16 + ho);
      o1 = wmma_bf16(pa, vb, o1);
      const __bf16* vp2 = &Vt[(2 * 16 + l) * 224 + k0];
      load_frag(vb, vp2 + ho, vp2 + 16 + ho);
      o2 = wmma_bf16(pa, vb, o2);
      const __bf16* vp3 = &Vt[(3 * 16 + l) * 224 + k0];
      load_frag(vb, vp3 + ho, vp3 + 16 + ho);
      o3 = wmma_bf16(pa, vb, o3);
    }
    #pragma unroll
    for (int r = 0; r < 8; ++r) {
      int t = rt * 16 + r + half * 8;
      if (t >= NW) continue;
      float ri = rinv[r];
      __bf16* op = attn + (tokBase + t) * C_ + head * HD + l;
      op[0]  = f2bf(o0[r] * ri);
      op[16] = f2bf(o1[r] * ri);
      op[32] = f2bf(o2[r] * ri);
      op[48] = f2bf(o3[r] * ri);
    }
  }
}

extern "C" void kernel_launch(void* const* d_in, const int* in_sizes, int n_in,
                              void* d_out, int out_size, void* d_ws, size_t ws_size,
                              hipStream_t stream) {
  (void)in_sizes; (void)n_in; (void)out_size; (void)ws_size;
  const float* x      = (const float*)d_in[0];
  const float* w_qkv  = (const float*)d_in[1];
  const float* b_qkv  = (const float*)d_in[2];
  const float* w_proj = (const float*)d_in[3];
  const float* b_proj = (const float*)d_in[4];
  const float* ln1_g  = (const float*)d_in[5];
  const float* ln1_b  = (const float*)d_in[6];
  const float* ln2_g  = (const float*)d_in[7];
  const float* ln2_b  = (const float*)d_in[8];
  const float* w_fc1  = (const float*)d_in[9];
  const float* b_fc1  = (const float*)d_in[10];
  const float* w_fc2  = (const float*)d_in[11];
  const float* b_fc2  = (const float*)d_in[12];

  char* ws = (char*)d_ws;
  size_t off = 0;
  auto take = [&](size_t bytes) {
    void* p = ws + off;
    off = (off + bytes + 255) & ~(size_t)255;
    return p;
  };
  __bf16* xn    = (__bf16*)take((size_t)TOKW * C_ * 2);
  __bf16* wqkvT = (__bf16*)take((size_t)N_QKV * C_ * 2);
  __bf16* wprjT = (__bf16*)take((size_t)C_ * C_ * 2);
  __bf16* wfc1T = (__bf16*)take((size_t)N_MLP * C_ * 2);
  __bf16* wfc2T = (__bf16*)take((size_t)C_ * N_MLP * 2);
  __bf16* qkvb  = (__bf16*)take((size_t)TOKW * N_QKV * 2);
  __bf16* attnb = (__bf16*)take((size_t)TOKW * C_ * 2);
  float*  x1    = (float*) take((size_t)TOKI * C_ * 4);
  __bf16* y2    = (__bf16*)take((size_t)TOKI * C_ * 2);
  __bf16* hmid  = (__bf16*)take((size_t)TOKI * N_MLP * 2);

  // weights -> bf16, transposed to [N][K]
  long t1 = (long)C_ * N_QKV;
  castT_kernel<<<(unsigned)((t1 + 255) / 256), 256, 0, stream>>>(w_qkv, wqkvT, C_, N_QKV);
  long t2 = (long)C_ * C_;
  castT_kernel<<<(unsigned)((t2 + 255) / 256), 256, 0, stream>>>(w_proj, wprjT, C_, C_);
  long t3 = (long)C_ * N_MLP;
  castT_kernel<<<(unsigned)((t3 + 255) / 256), 256, 0, stream>>>(w_fc1, wfc1T, C_, N_MLP);
  castT_kernel<<<(unsigned)((t3 + 255) / 256), 256, 0, stream>>>(w_fc2, wfc2T, N_MLP, C_);

  // LN1 directly into window-token order (zero rows for padding)
  ln_kernel<true><<<(TOKW + 7) / 8, 256, 0, stream>>>(x, ln1_g, ln1_b, xn, TOKW);

  // QKV projection
  gemm_kernel<0><<<dim3(N_QKV / 256, (TOKW + 63) / 64), 256, 0, stream>>>(
      xn, wqkvT, b_qkv, nullptr, qkvb, TOKW, N_QKV, C_);

  // windowed attention
  attn_kernel<<<dim3(NWIN, NHD), 256, 0, stream>>>(qkvb, attnb);

  // output projection + residual, scatter back to image layout (x1)
  gemm_kernel<1><<<dim3(C_ / 256, (TOKW + 63) / 64), 256, 0, stream>>>(
      attnb, wprjT, b_proj, x, x1, TOKW, C_, C_);

  // LN2
  ln_kernel<false><<<TOKI / 8, 256, 0, stream>>>(x1, ln2_g, ln2_b, y2, TOKI);

  // FC1 + erf-GELU
  gemm_kernel<2><<<dim3(N_MLP / 256, TOKI / 64), 256, 0, stream>>>(
      y2, wfc1T, b_fc1, nullptr, hmid, TOKI, N_MLP, C_);

  // FC2 + residual -> final output
  gemm_kernel<3><<<dim3(C_ / 256, TOKI / 64), 256, 0, stream>>>(
      hmid, wfc2T, b_fc2, x1, d_out, TOKI, C_, N_MLP);
}
